// InteractionNetwork_19816979104464
// MI455X (gfx1250) — matrix operand order repaired and checked by reading, not compiled, not run on hardware
//
#include <hip/hip_runtime.h>
#include <stdint.h>

typedef __bf16 bf16;
typedef __attribute__((ext_vector_type(16))) __bf16 bf16x16;
typedef __attribute__((ext_vector_type(8))) float v8f;

#define NB 16
#define NO 128
#define NR 16256
#define OD 64
#define RD 32
#define RT_PER_B (NR / 16) /* 1016 */
#define LDW 256            /* bf16 per LDS activation row (128 dwords), XOR swizzled */
#define KSPLIT 8

// ---------------- device helpers ----------------

__device__ __forceinline__ int swz(int m, int dw) { return dw ^ ((m & 15) << 2); }

__device__ __forceinline__ uint32_t pack2(float a, float b) {
  union { bf16 h[2]; uint32_t u; } p;
  p.h[0] = (bf16)a; p.h[1] = (bf16)b;
  return p.u;
}

__device__ __forceinline__ v8f wmma_bf16(bf16x16 a, bf16x16 b, v8f c) {
  return __builtin_amdgcn_wmma_f32_16x16x32_bf16(false, a, false, b, (short)0, c, false, false);
}

// ---- CDNA5 async global->LDS copy (16B per lane), tracked with ASYNCcnt ----
__device__ __forceinline__ void async_copy_b128(const void* gsrc, void* ldst) {
  const uint64_t g = (uint64_t)(uintptr_t)gsrc;
  const uint32_t l = (uint32_t)(uintptr_t)ldst;  // low 32 bits = LDS offset
  asm volatile("global_load_async_to_lds_b128 %0, %1, off" ::"v"(l), "v"(g) : "memory");
}
__device__ __forceinline__ void wait_async0() {
  asm volatile("s_wait_asynccnt 0" ::: "memory");
}

// issue async staging of rows [row0,row1) of a bf16 matrix [.][ldk] into LDS
// at row pitch ldp (absolute row indexing). Issue-only; caller waits + barriers.
__device__ __forceinline__ void stage_rows(bf16* dst, const bf16* src, int row0, int row1,
                                           int ldk, int ldp) {
  const int chunks = ldk >> 3;  // uint4 (8 bf16) per row
  const int total = (row1 - row0) * chunks;
  for (int i = threadIdx.x; i < total; i += 256) {
    const int r = row0 + i / chunks, c = i % chunks;
    async_copy_b128(src + (size_t)r * ldk + c * 8, dst + (size_t)r * ldp + c * 8);
  }
}

// A fragment (16x32) from swizzled LDS activation tile (16 rows x LDW bf16)
__device__ __forceinline__ bf16x16 ldsA(const bf16* tile, int kbase, int lane) {
  const int half = (lane >> 4) & 1, m = lane & 15;
  const uint32_t* rp = (const uint32_t*)(tile + m * LDW);
  union { bf16x16 v; uint32_t u[8]; } f;
#pragma unroll
  for (int i = 0; i < 8; ++i) {
    int k = 2 * i + 16 * half + kbase;  // contiguous k = 16*half + s (matches B)
    f.u[i] = rp[swz(m, k >> 1)];
  }
  return f.v;
}

// B fragment (32x16) from row-major [N][ld] bf16 matrix (K contiguous; global or LDS)
__device__ __forceinline__ bf16x16 gB(const bf16* W, int ld, int n0, int kbase, int lane) {
  const int half = (lane >> 4) & 1, n = n0 + (lane & 15);
  const uint32_t* p = (const uint32_t*)(W + (size_t)n * ld + kbase + 16 * half);
  union { bf16x16 v; uint32_t u[8]; } f;
#pragma unroll
  for (int i = 0; i < 8; ++i) f.u[i] = p[i];
  return f.v;
}

// guarded B fragment: columns >= nvalid are zero
__device__ __forceinline__ bf16x16 gBg(const bf16* W, int ld, int kbase, int lane, int nvalid) {
  const int half = (lane >> 4) & 1, n = lane & 15;
  const int nr = n < nvalid ? n : (nvalid - 1);
  const uint32_t* p = (const uint32_t*)(W + (size_t)nr * ld + kbase + 16 * half);
  union { bf16x16 v; uint32_t u[8]; } f;
#pragma unroll
  for (int i = 0; i < 8; ++i) f.u[i] = (n < nvalid) ? p[i] : 0u;
  return f.v;
}

// B fragment from row-major LDS slab [K=32][ldr] (lane walks its column n)
__device__ __forceinline__ bf16x16 ldsBrm(const bf16* slab, int ldr, int n0, int lane) {
  const int half = (lane >> 4) & 1, n = n0 + (lane & 15);
  union { bf16x16 v; bf16 h[16]; } f;
#pragma unroll
  for (int j = 0; j < 16; ++j) f.h[j] = slab[(j + 16 * half) * ldr + n];
  return f.v;
}

// A fragment from strided f32 relation matrix: element (m,k) = Sb[k*NR + r0+m]
__device__ __forceinline__ bf16x16 relA(const float* Sb, int r0, int kbase, int lane) {
  const int half = (lane >> 4) & 1, m = lane & 15;
  union { bf16x16 v; uint32_t u[8]; } f;
#pragma unroll
  for (int i = 0; i < 8; ++i) {
    int k = 2 * i + 16 * half + kbase;
    float a = Sb[(size_t)k * NR + r0 + m];
    float b = Sb[(size_t)(k + 1) * NR + r0 + m];
    f.u[i] = pack2(a, b);
  }
  return f.v;
}

// A fragment from this lane's contiguous f32 row (K contiguous)
__device__ __forceinline__ bf16x16 rowA(const float* rowp, int kbase, int lane) {
  const int half = (lane >> 4) & 1;
  union { bf16x16 v; uint32_t u[8]; } f;
#pragma unroll
  for (int i = 0; i < 8; ++i) {
    int k = 2 * i + 16 * half + kbase;
    float2 q = *(const float2*)(rowp + k);
    f.u[i] = pack2(q.x, q.y);
  }
  return f.v;
}

// store C tile into swizzled LDS activation tile as bf16
__device__ __forceinline__ void stC(bf16* tile, int n0, const v8f& acc, int lane) {
  const int half = (lane >> 4) & 1, n = n0 + (lane & 15);
#pragma unroll
  for (int vi = 0; vi < 8; ++vi) {
    int m = vi + 8 * half;
    tile[m * LDW + swz(m, n >> 1) * 2 + (n & 1)] = (bf16)acc[vi];
  }
}

// ---------------- small prep kernels ----------------

__global__ void k_f32_to_bf16(const float* __restrict__ s, bf16* __restrict__ d, int n) {
  const int i = blockIdx.x * blockDim.x + threadIdx.x;
  if (i < n) d[i] = (bf16)s[i];
}

__global__ void k_zero(float* __restrict__ p, int n) {
  const int i = blockIdx.x * blockDim.x + threadIdx.x;
  if (i < n) p[i] = 0.f;
}

__global__ void k_obT(const float* __restrict__ obj, bf16* __restrict__ obT) {
  const int i = blockIdx.x * blockDim.x + threadIdx.x;
  if (i >= NB * OD * NO) return;
  const int o = i % NO, d = (i / NO) % OD, b = i / (NO * OD);
  obT[i] = (bf16)obj[((size_t)b * NO + o) * OD + d];
}

__global__ void k_ybuild(const float* __restrict__ obj, const float* __restrict__ effrecv,
                         bf16* __restrict__ Y) {
  const int i = blockIdx.x * blockDim.x + threadIdx.x;
  if (i >= NB * NO * 192) return;
  const int c = i % 192, row = i / 192;
  const float v = (c < OD) ? obj[(size_t)row * OD + c] : effrecv[(size_t)row * 128 + (c - OD)];
  Y[i] = (bf16)v;
}

// ---------------- fused gather + relation MLP ----------------
// grid 2032 x 256 threads (8 waves, one 16-row tile each)
// dyn LDS: 16 activation tiles (128KB) + padded weight buffer (132KB) = 260KB of the
// WGP's 320KB. Weights staged per layer via async global->LDS, software-pipelined:
// half0 staged behind previous compute (gather for layer0), half1 behind pass0.

#define WBUF_OFF (16 * 16 * LDW) /* bf16 elements before weight buffer */

__global__ void k_relation(const float* __restrict__ S, const float* __restrict__ R,
                           const float* __restrict__ RI, const bf16* __restrict__ obT,
                           const bf16* __restrict__ rw0, const float* __restrict__ rb0,
                           const bf16* __restrict__ rw1, const float* __restrict__ rb1,
                           const bf16* __restrict__ rw2, const float* __restrict__ rb2,
                           bf16* __restrict__ effects) {
  extern __shared__ bf16 lds[];
  const int lane = threadIdx.x & 31;
  const int wid = threadIdx.x >> 5;  // 0..7
  const int half = (lane >> 4) & 1;
  bf16* bufA = lds + (size_t)wid * 16 * LDW;
  bf16* bufB = lds + (size_t)(8 + wid) * 16 * LDW;
  bf16* wbuf = lds + (size_t)WBUF_OFF;
  const int t = blockIdx.x * 8 + wid;  // 0..16255 row tiles (never straddle a batch)
  const int b = t / RT_PER_B;
  const int r0 = (t % RT_PER_B) * 16;
  const float* Sb = S + (size_t)b * NO * NR;
  const float* Rb = R + (size_t)b * NO * NR;
  const bf16* oTb = obT + (size_t)b * OD * NO;

  // kick off async staging of layer-0 weight half0; overlap with the gather GEMM
  stage_rows(wbuf, rw0, 0, 128, 160, 168);

  // ---- gather via WMMA: senders -> cols [0,64), receivers -> cols [64,128)
  {
    v8f acc[8] = {};
#pragma unroll
    for (int ks = 0; ks < 4; ++ks) {
      bf16x16 aS = relA(Sb, r0, ks * 32, lane);
      bf16x16 aR = relA(Rb, r0, ks * 32, lane);
#pragma unroll
      for (int nt = 0; nt < 4; ++nt) {
        bf16x16 bb = gB(oTb, NO, nt * 16, ks * 32, lane);
        acc[nt] = wmma_bf16(aS, bb, acc[nt]);
        acc[4 + nt] = wmma_bf16(aR, bb, acc[4 + nt]);
      }
    }
#pragma unroll
    for (int nt = 0; nt < 4; ++nt) {
      stC(bufA, nt * 16, acc[nt], lane);
      stC(bufA, 64 + nt * 16, acc[4 + nt], lane);
    }
  }
  // ---- relation_info -> cols [128,160)
  {
    const int c = 128 + lane;
#pragma unroll
    for (int m = 0; m < 16; ++m) {
      float v = RI[((size_t)b * NR + r0 + m) * RD + lane];
      bufA[m * LDW + swz(m, c >> 1) * 2 + (c & 1)] = (bf16)v;
    }
  }
  wait_async0();
  __syncthreads();

  // ---- layer0: 160 -> 256, relu (bufA -> bufB), weights in LDS (pitch 168)
#pragma unroll
  for (int pass = 0; pass < 2; ++pass) {
    if (pass == 0) stage_rows(wbuf, rw0, 128, 256, 160, 168);  // half1 behind pass0
    v8f acc[8] = {};
    for (int ks = 0; ks < 5; ++ks) {
      bf16x16 a = ldsA(bufA, ks * 32, lane);
#pragma unroll
      for (int nt = 0; nt < 8; ++nt)
        acc[nt] = wmma_bf16(a, gB(wbuf, 168, (pass * 8 + nt) * 16, ks * 32, lane), acc[nt]);
    }
#pragma unroll
    for (int nt = 0; nt < 8; ++nt) {
      const float bias = rb0[(pass * 8 + nt) * 16 + (lane & 15)];
      v8f tmp;
#pragma unroll
      for (int vi = 0; vi < 8; ++vi) tmp[vi] = fmaxf(acc[nt][vi] + bias, 0.f);
      stC(bufB, (pass * 8 + nt) * 16, tmp, lane);
    }
    wait_async0();
    __syncthreads();
  }

  // ---- layer1: 256 -> 256, relu (bufB -> bufA), weights in LDS (pitch 264)
  stage_rows(wbuf, rw1, 0, 128, 256, 264);
  wait_async0();
  __syncthreads();
#pragma unroll
  for (int pass = 0; pass < 2; ++pass) {
    if (pass == 0) stage_rows(wbuf, rw1, 128, 256, 256, 264);
    v8f acc[8] = {};
    for (int ks = 0; ks < 8; ++ks) {
      bf16x16 a = ldsA(bufB, ks * 32, lane);
#pragma unroll
      for (int nt = 0; nt < 8; ++nt)
        acc[nt] = wmma_bf16(a, gB(wbuf, 264, (pass * 8 + nt) * 16, ks * 32, lane), acc[nt]);
    }
#pragma unroll
    for (int nt = 0; nt < 8; ++nt) {
      const float bias = rb1[(pass * 8 + nt) * 16 + (lane & 15)];
      v8f tmp;
#pragma unroll
      for (int vi = 0; vi < 8; ++vi) tmp[vi] = fmaxf(acc[nt][vi] + bias, 0.f);
      stC(bufA, (pass * 8 + nt) * 16, tmp, lane);
    }
    wait_async0();
    __syncthreads();
  }

  // ---- layer2: 256 -> 128, relu -> effects (bf16 global), weights in LDS
  stage_rows(wbuf, rw2, 0, 64, 256, 264);
  wait_async0();
  __syncthreads();
#pragma unroll
  for (int pass = 0; pass < 2; ++pass) {
    if (pass == 0) stage_rows(wbuf, rw2, 64, 128, 256, 264);
    v8f acc[4] = {};
    for (int ks = 0; ks < 8; ++ks) {
      bf16x16 a = ldsA(bufA, ks * 32, lane);
#pragma unroll
      for (int nt = 0; nt < 4; ++nt)
        acc[nt] = wmma_bf16(a, gB(wbuf, 264, (pass * 4 + nt) * 16, ks * 32, lane), acc[nt]);
    }
#pragma unroll
    for (int nt = 0; nt < 4; ++nt) {
      const int n = (pass * 4 + nt) * 16 + (lane & 15);
      const float bias = rb2[n];
#pragma unroll
      for (int vi = 0; vi < 8; ++vi) {
        const int m = vi + 8 * half;
        effects[((size_t)b * NR + r0 + m) * 128 + n] = (bf16)fmaxf(acc[nt][vi] + bias, 0.f);
      }
    }
    if (pass == 0) {
      wait_async0();
      __syncthreads();
    }
  }
}

// ---------------- scatter GEMM: eff_recv += RR * effects ----------------
// grid = NB*KSPLIT blocks x 256 threads (8 waves; wave = one 16-row M tile)
// effects K-slab double-buffered via async global->LDS: copy of slab i+1 overlaps
// the 64 WMMAs of slab i.

#define SLDR 136 /* slab row pitch in bf16 */

__device__ __forceinline__ void issue_slab(const bf16* Eb, int r0, bf16* dst) {
  for (int i = threadIdx.x; i < 512; i += 256) {  // 32 rows x 16 x (8 bf16)
    const int rl = i >> 4, dq = i & 15;
    async_copy_b128(Eb + (size_t)(r0 + rl) * 128 + dq * 8, dst + rl * SLDR + dq * 8);
  }
}

__global__ void k_scatter(const float* __restrict__ R, const bf16* __restrict__ effects,
                          float* __restrict__ effrecv) {
  __shared__ bf16 slab[2][32 * SLDR];
  const int b = blockIdx.x / KSPLIT;
  const int c = blockIdx.x % KSPLIT;
  const int lane = threadIdx.x & 31;
  const int wid = threadIdx.x >> 5;
  const int half = (lane >> 4) & 1;
  const int KS_TOTAL = NR / 32;  // 508
  const int kchunk = (KS_TOTAL + KSPLIT - 1) / KSPLIT;
  const int ks0 = c * kchunk;
  const int ks1 = (ks0 + kchunk < KS_TOTAL) ? (ks0 + kchunk) : KS_TOTAL;
  const float* Rb = R + (size_t)b * NO * NR;
  const bf16* Eb = effects + (size_t)b * NR * 128;
  const float* arow = Rb + (size_t)(wid * 16 + (lane & 15)) * NR;

  int p = 0;
  issue_slab(Eb, ks0 * 32, slab[0]);
  wait_async0();
  __syncthreads();

  v8f acc[8] = {};
  for (int ks = ks0; ks < ks1; ++ks) {
    if (ks + 1 < ks1) {
      issue_slab(Eb, (ks + 1) * 32, slab[p ^ 1]);  // overlap copy with compute
      __builtin_prefetch((const void*)(arow + (size_t)(ks + 1) * 32), 0, 0);
    }
    bf16x16 a = rowA(arow, ks * 32, lane);
#pragma unroll
    for (int nt = 0; nt < 8; ++nt)
      acc[nt] = wmma_bf16(a, ldsBrm(slab[p], SLDR, nt * 16, lane), acc[nt]);
    wait_async0();
    __syncthreads();  // next slab complete AND all waves done with slab[p]
    p ^= 1;
  }
#pragma unroll
  for (int nt = 0; nt < 8; ++nt) {
    const int n = nt * 16 + (lane & 15);
#pragma unroll
    for (int vi = 0; vi < 8; ++vi) {
      const int o = wid * 16 + vi + 8 * half;
      atomicAdd(&effrecv[((size_t)b * NO + o) * 128 + n], acc[nt][vi]);
    }
  }
}

// ---------------- object MLP: 192 -> 256 -> 128 -> 3 ----------------
// grid 32 x 128 threads (4 waves), dyn LDS 64KB (weights stay in L2; tiny kernel)

__global__ void k_object(const bf16* __restrict__ Y, const bf16* __restrict__ ow0,
                         const float* __restrict__ ob0, const bf16* __restrict__ ow1,
                         const float* __restrict__ ob1, const bf16* __restrict__ ow2,
                         const float* __restrict__ ob2, float* __restrict__ out) {
  extern __shared__ bf16 lds[];
  const int lane = threadIdx.x & 31;
  const int wid = threadIdx.x >> 5;
  const int half = (lane >> 4) & 1;
  bf16* bufA = lds + (size_t)wid * 16 * LDW;
  bf16* bufB = lds + (size_t)(4 + wid) * 16 * LDW;
  const int t = blockIdx.x * 4 + wid;  // 0..127
  const int row0 = t * 16;

  // stage Y tile (16 rows x 192 bf16) into swizzled bufA (uint4 chunks)
  for (int i = lane; i < 16 * 24; i += 32) {
    const int m = i / 24, q = i % 24;
    const int sdw = (4 * q) ^ ((m & 15) << 2);
    *(uint4*)(bufA + m * LDW + sdw * 2) = *(const uint4*)(Y + (size_t)(row0 + m) * 192 + q * 8);
  }
  __syncthreads();

  // layer0: 192 -> 256, relu
#pragma unroll
  for (int pass = 0; pass < 2; ++pass) {
    v8f acc[8] = {};
    for (int ks = 0; ks < 6; ++ks) {
      bf16x16 a = ldsA(bufA, ks * 32, lane);
#pragma unroll
      for (int nt = 0; nt < 8; ++nt)
        acc[nt] = wmma_bf16(a, gB(ow0, 192, (pass * 8 + nt) * 16, ks * 32, lane), acc[nt]);
    }
#pragma unroll
    for (int nt = 0; nt < 8; ++nt) {
      const float bias = ob0[(pass * 8 + nt) * 16 + (lane & 15)];
      v8f tmp;
#pragma unroll
      for (int vi = 0; vi < 8; ++vi) tmp[vi] = fmaxf(acc[nt][vi] + bias, 0.f);
      stC(bufB, (pass * 8 + nt) * 16, tmp, lane);
    }
  }
  __syncthreads();

  // layer1: 256 -> 128, relu
  {
    v8f acc[8] = {};
    for (int ks = 0; ks < 8; ++ks) {
      bf16x16 a = ldsA(bufB, ks * 32, lane);
#pragma unroll
      for (int nt = 0; nt < 8; ++nt)
        acc[nt] = wmma_bf16(a, gB(ow1, 256, nt * 16, ks * 32, lane), acc[nt]);
    }
#pragma unroll
    for (int nt = 0; nt < 8; ++nt) {
      const float bias = ob1[nt * 16 + (lane & 15)];
      v8f tmp;
#pragma unroll
      for (int vi = 0; vi < 8; ++vi) tmp[vi] = fmaxf(acc[nt][vi] + bias, 0.f);
      stC(bufA, nt * 16, tmp, lane);
    }
  }
  __syncthreads();

  // layer2: 128 -> 3 (one guarded N tile), no relu
  {
    v8f acc = {};
    for (int ks = 0; ks < 4; ++ks)
      acc = wmma_bf16(ldsA(bufA, ks * 32, lane), gBg(ow2, 128, ks * 32, lane, 3), acc);
    const int n = lane & 15;
    if (n < 3) {
      const float bias = ob2[n];
#pragma unroll
      for (int vi = 0; vi < 8; ++vi)
        out[(size_t)(row0 + vi + 8 * half) * 3 + n] = acc[vi] + bias;
    }
  }
}

// ---------------- host launcher ----------------

extern "C" void kernel_launch(void* const* d_in, const int* in_sizes, int n_in,
                              void* d_out, int out_size, void* d_ws, size_t ws_size,
                              hipStream_t stream) {
  (void)in_sizes; (void)n_in; (void)out_size; (void)ws_size;
  const float* objects = (const float*)d_in[0];
  const float* S = (const float*)d_in[1];
  const float* R = (const float*)d_in[2];
  const float* RI = (const float*)d_in[3];
  const float* rW0 = (const float*)d_in[4];  const float* rb0 = (const float*)d_in[5];
  const float* rW1 = (const float*)d_in[6];  const float* rb1 = (const float*)d_in[7];
  const float* rW2 = (const float*)d_in[8];  const float* rb2 = (const float*)d_in[9];
  const float* oW0 = (const float*)d_in[10]; const float* ob0 = (const float*)d_in[11];
  const float* oW1 = (const float*)d_in[12]; const float* ob1 = (const float*)d_in[13];
  const float* oW2 = (const float*)d_in[14]; const float* ob2 = (const float*)d_in[15];
  float* out = (float*)d_out;

  char* ws = (char*)d_ws;
  size_t off = 0;
  auto alloc = [&](size_t bytes) -> void* {
    void* p = ws + off;
    off = (off + bytes + 255) & ~(size_t)255;
    return p;
  };
  bf16* effects = (bf16*)alloc((size_t)NB * NR * 128 * 2);  // 66.6 MB
  float* eff_recv = (float*)alloc((size_t)NB * NO * 128 * 4);
  bf16* obT = (bf16*)alloc((size_t)NB * OD * NO * 2);
  bf16* Y = (bf16*)alloc((size_t)NB * NO * 192 * 2);
  bf16* rw0b = (bf16*)alloc(256 * 160 * 2);
  bf16* rw1b = (bf16*)alloc(256 * 256 * 2);
  bf16* rw2b = (bf16*)alloc(128 * 256 * 2);
  bf16* ow0b = (bf16*)alloc(256 * 192 * 2);
  bf16* ow1b = (bf16*)alloc(128 * 256 * 2);
  bf16* ow2b = (bf16*)alloc(3 * 128 * 2);

  auto cvt = [&](const float* s, bf16* d, int n) {
    k_f32_to_bf16<<<(n + 255) / 256, 256, 0, stream>>>(s, d, n);
  };
  cvt(rW0, rw0b, 256 * 160);
  cvt(rW1, rw1b, 256 * 256);
  cvt(rW2, rw2b, 128 * 256);
  cvt(oW0, ow0b, 256 * 192);
  cvt(oW1, ow1b, 128 * 256);
  cvt(oW2, ow2b, 3 * 128);
  k_obT<<<(NB * OD * NO + 255) / 256, 256, 0, stream>>>(objects, obT);
  k_zero<<<(NB * NO * 128 + 255) / 256, 256, 0, stream>>>(eff_recv, NB * NO * 128);

  // relation kernel: 8 waves/WG, 260KB LDS (16 act tiles + 264-pitch weight buffer)
  const size_t relLds = ((size_t)WBUF_OFF + 256 * 264) * 2;
  k_relation<<<(NB * RT_PER_B) / 8, 256, relLds, stream>>>(
      S, R, RI, obT, rw0b, rb0, rw1b, rb1, rw2b, rb2, effects);

  k_scatter<<<NB * KSPLIT, 256, 0, stream>>>(R, effects, eff_recv);

  k_ybuild<<<(NB * NO * 192 + 255) / 256, 256, 0, stream>>>(objects, eff_recv, Y);

  const size_t objLds = (size_t)8 * 16 * LDW * 2;  // 64 KB
  k_object<<<(NB * NO / 16) / 4, 128, objLds, stream>>>(
      Y, ow0b, ob0, ow1b, ob1, ow2b, ob2, out);
}
